// VanillaRNN_71528385347793
// MI455X (gfx1250) — matrix operand order repaired and checked by reading, not compiled, not run on hardware
//
#include <hip/hip_runtime.h>

// ---------------------------------------------------------------------------
// Vanilla RNN forward for MI455X (gfx1250, wave32, WMMA).
//   B=256, T=512, D=128, H=512, C=128
// Prep:    x -> bf16; Whx/Whh/Wph -> bf16 repacked in WMMA B-fragment order
// Phase 1: xproj[t][b][h] = x@Whx + bh   (bf16 WMMA, f32 accum, stored f32)
// Phase 2: persistent scan, 16 WGs x 16 batch rows, h in LDS bf16,
//          Whh fragments streamed from L2, next-step xproj slab prefetched
//          (global_prefetch_b8) to keep HBM latency off the recurrence chain
// Phase 3: out = h_T @ Wph + bp          (bf16 WMMA)
// ---------------------------------------------------------------------------

typedef __bf16 bf16_t;
typedef __attribute__((ext_vector_type(16))) __bf16 v16bf;
typedef __attribute__((ext_vector_type(8)))  __bf16 v8bf;
typedef __attribute__((ext_vector_type(8)))  float  v8f;

#define WMMA_BF16(a, b, c) \
  __builtin_amdgcn_wmma_f32_16x16x32_bf16(false, (a), false, (b), (short)0, (c), false, false)

static constexpr int BB = 256;   // batch
static constexpr int TT = 512;   // time
static constexpr int DD = 128;   // input dim
static constexpr int HH = 512;   // hidden
static constexpr int CC = 128;   // output classes
static constexpr int HP = HH + 8; // padded LDS row (1040 B: 16B-aligned, conflict-free)

// float -> bf16 RNE (bit-level; only used in prep kernels / scan epilogue)
static __device__ __forceinline__ bf16_t f2bf(float f) {
  unsigned int u = __builtin_bit_cast(unsigned int, f);
  unsigned int r = u + 0x7FFFu + ((u >> 16) & 1u);
  unsigned short h = (unsigned short)(r >> 16);
  return __builtin_bit_cast(bf16_t, h);
}

static __device__ __forceinline__ v16bf concat8(v8bf lo, v8bf hi) {
  return __builtin_shufflevector(lo, hi, 0, 1, 2, 3, 4, 5, 6, 7,
                                         8, 9, 10, 11, 12, 13, 14, 15);
}

// A-fragment (16x32 bf16): per lane, K = k0 + hi8 + {0..7} and k0 + 16 + hi8 + {0..7}.
// p must point at (row, k0 + hi8), 16B aligned. Two contiguous 16B loads.
static __device__ __forceinline__ v16bf load_a_frag(const bf16_t* p) {
  v8bf lo = *(const v8bf*)p;
  v8bf hi = *(const v8bf*)(p + 16);
  return concat8(lo, hi);
}

// B-fragment from repacked weights: 512 contiguous bf16 per (ntile,kt) chunk,
// lane's 16 elements contiguous at chunk + lane*16. Two contiguous 16B loads.
static __device__ __forceinline__ v16bf load_b_frag(const bf16_t* chunk, int lane) {
  const bf16_t* p = chunk + lane * 16;
  v8bf lo = *(const v8bf*)p;
  v8bf hi = *(const v8bf*)(p + 8);
  return concat8(lo, hi);
}

// ---------------------------------------------------------------------------
// Prep A: f32 -> bf16, 8 elements/thread (b128 in, b128 out). Bandwidth-bound.
__global__ __launch_bounds__(256) void cvt8_f32_to_bf16_kernel(
    const float* __restrict__ in, bf16_t* __restrict__ out, int n8) {
  int i = blockIdx.x * 256 + threadIdx.x;
  if (i >= n8) return;
  const float* p = in + (size_t)i * 8;
  v8bf o;
#pragma unroll
  for (int j = 0; j < 8; ++j) o[j] = f2bf(p[j]);
  *(v8bf*)(out + (size_t)i * 8) = o;
}

// ---------------------------------------------------------------------------
// Prep B: repack a row-major [K x N] f32 matrix into bf16 WMMA B-fragment order.
// chunk c = ntile*(K/32) + kt holds 32 lanes x 16 elems:
//   elem(lane,e) = W[(kt*32 + (e&7) + 8*(lane>>4) + 16*(e>>3)) * N + ntile*16 + (lane&15)]
__global__ __launch_bounds__(256) void repack_b_kernel(
    const float* __restrict__ W, bf16_t* __restrict__ out, int K, int N) {
  int o = blockIdx.x * 256 + threadIdx.x;
  if (o >= K * N) return;
  const int c = o >> 9;
  const int r = o & 511;
  const int lane = r >> 4;
  const int e = r & 15;
  const int kchunks = K >> 5;
  const int ntile = c / kchunks;
  const int kt = c % kchunks;
  const int k = kt * 32 + (e & 7) + ((lane >> 4) & 1) * 8 + (e >> 3) * 16;
  const int n = ntile * 16 + (lane & 15);
  out[o] = f2bf(W[(size_t)k * N + n]);
}

// ---------------------------------------------------------------------------
// Phase 1: xproj GEMM. grid=8192, block=256 (8 waves).
// Each wave computes a 16(M) x 64(N) strip: 1 A-frag feeds 4 WMMAs per k step.
// Output stored time-major: xproj[t][b][h].
__global__ __launch_bounds__(256) void xproj_gemm_kernel(
    const bf16_t* __restrict__ xb,     // [B*T, D] bf16
    const bf16_t* __restrict__ WhxF,   // repacked fragments, kchunks = D/32 = 4
    const float* __restrict__ bh,      // [H]
    float* __restrict__ xproj) {       // [T,B,H] f32
  const int lane  = threadIdx.x & 31;
  const int wave  = threadIdx.x >> 5;
  const int mTile = blockIdx.x;                 // 0..8191 over M=B*T
  const int hi    = (lane >> 4) & 1;
  const int rlo   = lane & 15;
  const long gr   = (long)mTile * 16 + rlo;
  const bf16_t* arow = xb + gr * DD + hi * 8;
  const int nt0 = wave * 4;                     // 4 N-tiles per wave
  constexpr int KCH = DD / 32;                  // 4

  v8f acc[4] = {};
#pragma unroll
  for (int kt = 0; kt < KCH; ++kt) {
    const v16bf a = load_a_frag(arow + kt * 32);
#pragma unroll
    for (int j = 0; j < 4; ++j) {
      const v16bf b = load_b_frag(WhxF + (((size_t)(nt0 + j) * KCH + kt) << 9), lane);
      acc[j] = WMMA_BF16(a, b, acc[j]);
    }
  }
#pragma unroll
  for (int j = 0; j < 4; ++j) {
    const int n = (nt0 + j) * 16 + rlo;
    const float bias = bh[n];
#pragma unroll
    for (int v = 0; v < 8; ++v) {
      const int  m  = v + hi * 8;
      const long g  = (long)mTile * 16 + m;
      const int  bb = (int)(g >> 9);            // /T
      const int  tt = (int)(g & (TT - 1));      // %T
      xproj[((long)tt * BB + bb) * HH + n] = acc[j][v] + bias;
    }
  }
}

// ---------------------------------------------------------------------------
// Phase 2: persistent scan. grid=16 (B/16), block=512 (16 waves).
// Each wave owns 32 hidden columns (2 N-tiles, shared A-frag); h in LDS bf16.
// Next timestep's 32 KB xproj slab is prefetched (512 thr x 64 B) so the
// epilogue's HBM reads are cache hits; keeps the serial chain on WMMA+LDS.
__global__ __launch_bounds__(512) void rnn_scan_kernel(
    const float* __restrict__ xproj,   // [T,B,H] f32
    const bf16_t* __restrict__ WhhF,   // repacked fragments, kchunks = H/32 = 16
    bf16_t* __restrict__ hfin) {       // [B,H] bf16
  __shared__ __align__(16) bf16_t hbuf[2][16][HP];

  const int tid   = threadIdx.x;
  const int lane  = tid & 31;
  const int wave  = tid >> 5;                  // 0..15
  const int hi    = (lane >> 4) & 1;
  const int nlo   = lane & 15;
  const int bBase = blockIdx.x * 16;
  constexpr int KCH = HH / 32;                 // 16
  const int nt0 = wave * 2;
  const int n0 = nt0 * 16 + nlo;
  const int n1 = n0 + 16;

  for (int i = tid; i < 2 * 16 * HP; i += 512)
    ((bf16_t*)hbuf)[i] = f2bf(0.0f);

  // Warm the first xproj slab while waves converge on the first barrier.
  {
    const char* p0 = (const char*)(xproj + (long)bBase * HH) + tid * 64;
    __builtin_prefetch(p0, 0, 1);
  }
  __syncthreads();

  for (int t = 0; t < TT; ++t) {
    const int cur = t & 1;
    const int nxt = cur ^ 1;
    const float* xp = xproj + ((long)t * BB + bBase) * HH;
    const bf16_t* arow = &hbuf[cur][nlo][hi * 8];

    // Prefetch next step's slab; overlaps with this step's 32 WMMAs.
    if (t + 1 < TT) {
      const char* pn = (const char*)(xproj + ((long)(t + 1) * BB + bBase) * HH) + tid * 64;
      __builtin_prefetch(pn, 0, 1);
    }

    v8f acc0 = {}, acc1 = {};
#pragma unroll 4
    for (int kt = 0; kt < KCH; ++kt) {
      const v16bf a  = load_a_frag(arow + kt * 32);
      const v16bf b0 = load_b_frag(WhhF + (((size_t)nt0 * KCH + kt) << 9), lane);
      const v16bf b1 = load_b_frag(WhhF + ((((size_t)nt0 + 1) * KCH + kt) << 9), lane);
      acc0 = WMMA_BF16(a, b0, acc0);
      acc1 = WMMA_BF16(a, b1, acc1);
    }
#pragma unroll
    for (int v = 0; v < 8; ++v) {
      const int m = v + hi * 8;
      hbuf[nxt][m][n0] = f2bf(tanhf(acc0[v] + xp[m * HH + n0]));
      hbuf[nxt][m][n1] = f2bf(tanhf(acc1[v] + xp[m * HH + n1]));
    }
    __syncthreads();   // single barrier/step; double buffer makes it sufficient
  }

  // T even -> final state in buffer 0
  for (int i = tid; i < 16 * HH; i += 512) {
    const int m = i >> 9, n = i & (HH - 1);
    hfin[(long)(bBase + m) * HH + n] = hbuf[0][m][n];
  }
}

// ---------------------------------------------------------------------------
// Phase 3: out = h_T @ Wph + bp. grid=16, block=256 (8 waves), 128 tiles.
__global__ __launch_bounds__(256) void out_gemm_kernel(
    const bf16_t* __restrict__ hfin,   // [B,H] bf16
    const bf16_t* __restrict__ WphF,   // repacked fragments, kchunks = H/32 = 16
    const float* __restrict__ bp,      // [C]
    float* __restrict__ out) {         // [B,C] f32
  const int lane  = threadIdx.x & 31;
  const int wave  = threadIdx.x >> 5;
  const int tile  = blockIdx.x * 8 + wave;     // 0..127
  const int mTile = tile >> 3;                 // 16 tiles over M=256
  const int nTile = tile & 7;                  // 8 tiles over N=128
  const int hi    = (lane >> 4) & 1;
  const int rlo   = lane & 15;
  const int n     = nTile * 16 + rlo;
  const int r     = mTile * 16 + rlo;
  constexpr int KCH = HH / 32;                 // 16
  const bf16_t* arow = hfin + (size_t)r * HH + hi * 8;

  v8f acc = {};
#pragma unroll 4
  for (int kt = 0; kt < KCH; ++kt) {
    const v16bf a = load_a_frag(arow + kt * 32);
    const v16bf b = load_b_frag(WphF + (((size_t)nTile * KCH + kt) << 9), lane);
    acc = WMMA_BF16(a, b, acc);
  }
  const float bias = bp[n];
#pragma unroll
  for (int v = 0; v < 8; ++v) {
    const int m = mTile * 16 + v + hi * 8;
    out[(long)m * CC + n] = acc[v] + bias;
  }
}

// ---------------------------------------------------------------------------
extern "C" void kernel_launch(void* const* d_in, const int* in_sizes, int n_in,
                              void* d_out, int out_size, void* d_ws, size_t ws_size,
                              hipStream_t stream) {
  const float* x   = (const float*)d_in[0];   // [B,T,D]
  const float* Whx = (const float*)d_in[1];   // [D,H]
  const float* Whh = (const float*)d_in[2];   // [H,H]
  const float* Wph = (const float*)d_in[3];   // [H,C]
  const float* bh  = (const float*)d_in[4];   // [H]
  const float* bp  = (const float*)d_in[5];   // [C]

  char* ws = (char*)d_ws;
  size_t off = 0;
  float*  xproj = (float*)(ws + off);  off += (size_t)TT * BB * HH * sizeof(float);   // 256 MB
  bf16_t* xb    = (bf16_t*)(ws + off); off += (size_t)BB * TT * DD * sizeof(bf16_t); // 32 MB
  bf16_t* WhxF  = (bf16_t*)(ws + off); off += (size_t)DD * HH * sizeof(bf16_t);
  bf16_t* WhhF  = (bf16_t*)(ws + off); off += (size_t)HH * HH * sizeof(bf16_t);
  bf16_t* WphF  = (bf16_t*)(ws + off); off += (size_t)HH * CC * sizeof(bf16_t);
  bf16_t* hfin  = (bf16_t*)(ws + off); off += (size_t)BB * HH * sizeof(bf16_t);

  const int nx8 = (BB * TT * DD) / 8;   // 2,097,152
  cvt8_f32_to_bf16_kernel<<<(nx8 + 255) / 256, 256, 0, stream>>>(x, xb, nx8);
  repack_b_kernel<<<(DD * HH + 255) / 256, 256, 0, stream>>>(Whx, WhxF, DD, HH);
  repack_b_kernel<<<(HH * HH + 255) / 256, 256, 0, stream>>>(Whh, WhhF, HH, HH);
  repack_b_kernel<<<(HH * CC + 255) / 256, 256, 0, stream>>>(Wph, WphF, HH, CC);

  xproj_gemm_kernel<<<(BB * TT) / 16, 256, 0, stream>>>(xb, WhxF, bh, xproj);
  rnn_scan_kernel<<<BB / 16, 512, 0, stream>>>(xproj, WhhF, hfin);
  out_gemm_kernel<<<16, 256, 0, stream>>>(hfin, WphF, bp, (float*)d_out);
}
